// transformer_54082228191956
// MI455X (gfx1250) — compile-verified, matrix operand order checked by reading, not run on hardware
//
#include <hip/hip_runtime.h>
#include <math.h>

// ---------------- problem constants ----------------
#define DIMM   768
#define DEPTH  6
#define NHEAD  12
#define DHEAD  64
#define HD     768
#define PDIM   192
#define BB     2
#define NN     1024
#define MTOT   (BB * NN)        // 2048 rows in the residual stream
#define GLU_OUT (8 * DIMM)      // 6144
#define FF_HID  (4 * DIMM)      // 3072
#define NEGMAX 3.402823466e+38f

typedef __attribute__((ext_vector_type(16))) __bf16 bf16x16;
typedef __attribute__((ext_vector_type(8)))  float  floatx8;

__device__ __forceinline__ floatx8 wmma_bf16(bf16x16 a, bf16x16 b, floatx8 c) {
  // v_wmma_f32_16x16x32_bf16: D = A(16x32) * B(32x16) + C(16x16 f32)
  return __builtin_amdgcn_wmma_f32_16x16x32_bf16(false, a, false, b, (short)0, c,
                                                 false, false);
}

// Async global->LDS copy (gfx1250): data bypasses VGPRs, tracked by ASYNCcnt.
// Per-lane: copies 16 bytes from gptr to lds_ptr.
__device__ __forceinline__ void async_ld_b128(void* lds_ptr, const void* gptr) {
  unsigned l = (unsigned)(size_t)lds_ptr;  // generic->LDS offset (addr[31:0])
  asm volatile("global_load_async_to_lds_b128 %0, %1, off"
               :: "v"(l), "v"(gptr) : "memory");
}
__device__ __forceinline__ void wait_async0() {
  asm volatile("s_wait_asynccnt 0" ::: "memory");
}

// Load a 16x32 bf16 fragment from a row-major [16][>=32] region.
// Works for A-layout (row = M) and for B-layout when B is staged as [n][k].
// 16-bit layout: lane<16 holds row=lane, k = {0..7, 16..23};
//                lane>=16 holds row=lane-16, k = {8..15, 24..31}.
__device__ __forceinline__ bf16x16 ld_frag(const __bf16* p, int ld) {
  const int lane = threadIdx.x & 31;
  const int row  = lane & 15;
  const int hk   = (lane >> 4) << 3;
  const __bf16* r = p + row * ld;
  bf16x16 f;
#pragma unroll
  for (int e = 0; e < 8; ++e) f[e] = r[hk + e];
#pragma unroll
  for (int e = 0; e < 8; ++e) f[8 + e] = r[16 + hk + e];
  return f;
}

// ---------------- dynamic position bias MLP: htab[2N-1][NHEAD] ----------------
__global__ __launch_bounds__(PDIM) void posmlp_k(
    const float* __restrict__ pw1, const float* __restrict__ pb1,
    const float* __restrict__ pw2, const float* __restrict__ pb2,
    const float* __restrict__ pw3, const float* __restrict__ pb3,
    float* __restrict__ htab) {
  __shared__ float h1[PDIM];
  __shared__ float h2[PDIM];
  const int pidx = blockIdx.x;              // 0 .. 2N-2
  const float pos = (float)(pidx - (NN - 1));
  const int j = threadIdx.x;
  h1[j] = fmaxf(pos * pw1[j] + pb1[j], 0.f);
  __syncthreads();
  float acc = pb2[j];
  for (int k = 0; k < PDIM; ++k) acc += h1[k] * pw2[k * PDIM + j];
  h2[j] = fmaxf(acc, 0.f);
  __syncthreads();
  if (j < NHEAD) {
    float a = pb3[j];
    for (int k = 0; k < PDIM; ++k) a += h2[k] * pw3[k * NHEAD + j];
    htab[pidx * NHEAD + j] = a;
  }
}

// ---------------- per-batch temperature ----------------
__global__ __launch_bounds__(256) void temp_k(const unsigned char* __restrict__ mask,
                                              const float* __restrict__ temperature,
                                              float* __restrict__ tempv) {
  __shared__ int red[256];
  const int tid = threadIdx.x;
  for (int b = 0; b < BB; ++b) {
    int cnt = 0;
    for (int j = tid; j < NN; j += 256) cnt += mask[b * NN + j] ? 0 : 1;
    red[tid] = cnt;
    __syncthreads();
    for (int s = 128; s > 0; s >>= 1) {
      if (tid < s) red[tid] += red[tid + s];
      __syncthreads();
    }
    if (tid == 0) {
      float L = (float)red[0];
      tempv[b] = temperature[0] * log2f(L * L - L);
    }
    __syncthreads();
  }
}

// ---------------- LayerNorm -> bf16 ----------------
__global__ __launch_bounds__(256) void ln_k(const float* __restrict__ x,
                                            const float* __restrict__ g,
                                            const float* __restrict__ be,
                                            __bf16* __restrict__ out) {
  __shared__ float red[256];
  const int row = blockIdx.x, tid = threadIdx.x;
  const float* xp = x + (size_t)row * DIMM;
  float v0 = xp[tid], v1 = xp[tid + 256], v2 = xp[tid + 512];
  red[tid] = v0 + v1 + v2;
  __syncthreads();
  for (int s = 128; s > 0; s >>= 1) {
    if (tid < s) red[tid] += red[tid + s];
    __syncthreads();
  }
  const float mean = red[0] * (1.0f / DIMM);
  __syncthreads();
  float d0 = v0 - mean, d1 = v1 - mean, d2 = v2 - mean;
  red[tid] = d0 * d0 + d1 * d1 + d2 * d2;
  __syncthreads();
  for (int s = 128; s > 0; s >>= 1) {
    if (tid < s) red[tid] += red[tid + s];
    __syncthreads();
  }
  const float rstd = rsqrtf(red[0] * (1.0f / DIMM) + 1e-5f);
  __bf16* op = out + (size_t)row * DIMM;
  op[tid]       = (__bf16)(d0 * rstd * g[tid]       + be[tid]);
  op[tid + 256] = (__bf16)(d1 * rstd * g[tid + 256] + be[tid + 256]);
  op[tid + 512] = (__bf16)(d2 * rstd * g[tid + 512] + be[tid + 512]);
}

// ---------------- WMMA GEMM: C[M,N] = A_bf16[M,K] @ W_f32[K,N] (+bias)(+res) ----------------
// Block: 256 threads (8 waves), tile 128x64, K-step 32, double-buffered LDS.
// A tile staged with async global->LDS copies; W tile converted fp32->bf16 and
// stored transposed [n][k]. Wave (wm,wn) computes a 32x32 output (2x2 WMMA accs).
// EPI bit0 = add bias[N], bit1 = add residual R[M,N].
template <int EPI>
__global__ __launch_bounds__(256) void gemm_bf16_k(
    const __bf16* __restrict__ A, int lda, const float* __restrict__ W, int ldw,
    float* __restrict__ C, int ldc, const float* __restrict__ bias,
    const float* __restrict__ R, int K) {
  __shared__ __align__(16) __bf16 As[2][128][40];  // [m][k], +8 pad
  __shared__ __align__(16) __bf16 Ws[2][64][40];   // [n][k] transposed
  const int n0 = blockIdx.x * 64;
  const int m0 = blockIdx.y * 128;
  const int tid = threadIdx.x;
  const int wid = tid >> 5;
  const int wm = wid & 3, wn = wid >> 2;
  const int amr = tid >> 1, ac0 = (tid & 1) << 4;   // A stage: 128 rows x 32 k
  const int wkr = tid >> 3, wc0 = (tid & 7) << 3;   // W stage: 32 k x 64 n
  floatx8 acc[2][2] = {};
  const int nk = K >> 5;

  auto stage = [&](int buf, int k0) {
    const __bf16* ap = A + (size_t)(m0 + amr) * lda + k0 + ac0;
    async_ld_b128(&As[buf][amr][ac0], ap);
    async_ld_b128(&As[buf][amr][ac0 + 8], ap + 8);
    const float* wp = W + (size_t)(k0 + wkr) * ldw + n0 + wc0;
    float4 w0 = *(const float4*)wp;
    float4 w1 = *(const float4*)(wp + 4);
    Ws[buf][wc0 + 0][wkr] = (__bf16)w0.x; Ws[buf][wc0 + 1][wkr] = (__bf16)w0.y;
    Ws[buf][wc0 + 2][wkr] = (__bf16)w0.z; Ws[buf][wc0 + 3][wkr] = (__bf16)w0.w;
    Ws[buf][wc0 + 4][wkr] = (__bf16)w1.x; Ws[buf][wc0 + 5][wkr] = (__bf16)w1.y;
    Ws[buf][wc0 + 6][wkr] = (__bf16)w1.z; Ws[buf][wc0 + 7][wkr] = (__bf16)w1.w;
  };

  stage(0, 0);
  int buf = 0;
  for (int kt = 0; kt < nk; ++kt) {
    wait_async0();       // this wave's async tile writes are in LDS
    __syncthreads();     // all waves' tiles ready; prior reads of buf^1 done
    if (kt + 1 < nk) stage(buf ^ 1, (kt + 1) << 5);  // prefetch under compute
    bf16x16 a0 = ld_frag(&As[buf][wm * 32][0], 40);
    bf16x16 a1 = ld_frag(&As[buf][wm * 32 + 16][0], 40);
    bf16x16 b0 = ld_frag(&Ws[buf][wn * 32][0], 40);
    bf16x16 b1 = ld_frag(&Ws[buf][wn * 32 + 16][0], 40);
    acc[0][0] = wmma_bf16(a0, b0, acc[0][0]);
    acc[0][1] = wmma_bf16(a0, b1, acc[0][1]);
    acc[1][0] = wmma_bf16(a1, b0, acc[1][0]);
    acc[1][1] = wmma_bf16(a1, b1, acc[1][1]);
    buf ^= 1;
  }
  // C layout: lane<16 -> row r, lane>=16 -> row r+8; col = lane&15
  const int lane = tid & 31, nl = lane & 15, half = lane >> 4;
#pragma unroll
  for (int mi = 0; mi < 2; ++mi) {
#pragma unroll
    for (int ni = 0; ni < 2; ++ni) {
      const int col = n0 + wn * 32 + ni * 16 + nl;
      const float bv = (EPI & 1) ? bias[col] : 0.f;
#pragma unroll
      for (int r = 0; r < 8; ++r) {
        const int row = m0 + wm * 32 + mi * 16 + r + 8 * half;
        float v = acc[mi][ni][r] + bv;
        if (EPI & 2) v += R[(size_t)row * ldc + col];
        C[(size_t)row * ldc + col] = v;
      }
    }
  }
}

// ---------------- qkv de-interleave + l2norm(q,k) -> bf16 [B,H,N,64] ----------------
__global__ __launch_bounds__(384) void qkvprep_k(const float* __restrict__ qkv,
                                                 __bf16* __restrict__ qb,
                                                 __bf16* __restrict__ kb,
                                                 __bf16* __restrict__ vb) {
  const int row = blockIdx.x;               // b*N + n
  const int b = row >> 10, n = row & 1023;
  const int h = threadIdx.x >> 5, lane = threadIdx.x & 31;
  const int d0 = lane, d1 = lane + 32;
  const float* src = qkv + (size_t)row * (3 * HD) + h * (DHEAD * 3);
  float q0 = src[d0 * 3 + 0], q1 = src[d1 * 3 + 0];
  float k0 = src[d0 * 3 + 1], k1 = src[d1 * 3 + 1];
  float v0 = src[d0 * 3 + 2], v1 = src[d1 * 3 + 2];
  float sq = q0 * q0 + q1 * q1;
  float sk = k0 * k0 + k1 * k1;
  for (int off = 1; off < 32; off <<= 1) {
    sq += __shfl_xor(sq, off, 32);
    sk += __shfl_xor(sk, off, 32);
  }
  const float iq = 1.f / fmaxf(sqrtf(sq), 1e-12f);
  const float ik = 1.f / fmaxf(sqrtf(sk), 1e-12f);
  const size_t dst = ((size_t)(b * NHEAD + h) * NN + n) * DHEAD;
  qb[dst + d0] = (__bf16)(q0 * iq); qb[dst + d1] = (__bf16)(q1 * iq);
  kb[dst + d0] = (__bf16)(k0 * ik); kb[dst + d1] = (__bf16)(k1 * ik);
  vb[dst + d0] = (__bf16)v0;        vb[dst + d1] = (__bf16)v1;
}

// ---------------- flash-style causal cosine attention ----------------
// grid (N/64, B*H), block 128 (4 waves). Wave owns 16 q-rows x full d=64.
__global__ __launch_bounds__(128) void attn_k(const __bf16* __restrict__ qb,
                                              const __bf16* __restrict__ kb,
                                              const __bf16* __restrict__ vb,
                                              const unsigned char* __restrict__ mask,
                                              const float* __restrict__ tempv,
                                              const float* __restrict__ htab,
                                              __bf16* __restrict__ out) {
  __shared__ __align__(16) __bf16 ks[32][72];  // K chunk [j][d], async-staged
  __shared__ __bf16 vt[64][40];                // V chunk transposed [d][j]
  __shared__ __bf16 pt[4][16][40];             // per-wave P tile [m][j]
  const int bh = blockIdx.y, b = bh / NHEAD, h = bh % NHEAD;
  const size_t base = (size_t)bh * NN * DHEAD;
  const int tid = threadIdx.x, wid = tid >> 5, lane = tid & 31;
  const int nl = lane & 15, half = lane >> 4;
  const int qrow0 = blockIdx.x * 64 + wid * 16;

  const bf16x16 q_lo = ld_frag(qb + base + (size_t)qrow0 * DHEAD, DHEAD);
  const bf16x16 q_hi = ld_frag(qb + base + (size_t)qrow0 * DHEAD + 32, DHEAD);
  const float tscale = tempv[b];

  int qmask[8];
#pragma unroll
  for (int r = 0; r < 8; ++r) qmask[r] = mask[b * NN + qrow0 + r + 8 * half];

  float m_i[8], l_i[8];
  floatx8 o[4] = {};
#pragma unroll
  for (int r = 0; r < 8; ++r) { m_i[r] = -NEGMAX; l_i[r] = 0.f; }

  const int nch = 2 * blockIdx.x + 2;  // causal: only chunks with j <= block max row
  for (int c = 0; c < nch; ++c) {
    const int jc = c << 5;
    __syncthreads();
    {  // stage 32 keys (async direct-to-LDS) + values (transposed scatter)
      const int jr = tid >> 2, c0 = (tid & 3) << 4;
      const __bf16* kp = kb + base + (size_t)(jc + jr) * DHEAD + c0;
      async_ld_b128(&ks[jr][c0], kp);
      async_ld_b128(&ks[jr][c0 + 8], kp + 8);
      const __bf16* vp = vb + base + (size_t)(jc + jr) * DHEAD + c0;
#pragma unroll
      for (int u = 0; u < 16; ++u) vt[c0 + u][jr] = vp[u];
    }
    wait_async0();
    __syncthreads();

    // S = Q(16x64) . K_chunk(32x64)^T  -> two 16x16 tiles, K=64 chained as 2x32
    floatx8 s0 = {}, s1 = {};
    {
      bf16x16 blo = ld_frag(&ks[0][0], 72);
      bf16x16 bhi = ld_frag(&ks[0][32], 72);
      s0 = wmma_bf16(q_lo, blo, s0);
      s0 = wmma_bf16(q_hi, bhi, s0);
      blo = ld_frag(&ks[16][0], 72);
      bhi = ld_frag(&ks[16][32], 72);
      s1 = wmma_bf16(q_lo, blo, s1);
      s1 = wmma_bf16(q_hi, bhi, s1);
    }
    const int j0 = jc + nl, j1 = jc + 16 + nl;
    const int km0 = mask[b * NN + j0], km1 = mask[b * NN + j1];
#pragma unroll
    for (int r = 0; r < 8; ++r) {
      const int i = qrow0 + r + 8 * half;
      float v0, v1;
      if (j0 > i || qmask[r] || km0) v0 = -NEGMAX;
      else v0 = s0[r] * tscale + htab[(i - j0 + NN - 1) * NHEAD + h];
      if (j1 > i || qmask[r] || km1) v1 = -NEGMAX;
      else v1 = s1[r] * tscale + htab[(i - j1 + NN - 1) * NHEAD + h];
      s0[r] = v0;
      s1[r] = v1;
    }
    // online softmax; row stats: C-layout row r(+8*half) spread across 16 lanes
#pragma unroll
    for (int r = 0; r < 8; ++r) {
      float mx = fmaxf(s0[r], s1[r]);
      for (int off = 1; off < 16; off <<= 1) mx = fmaxf(mx, __shfl_xor(mx, off, 32));
      const float mn = fmaxf(m_i[r], mx);
      const float alpha = __expf(m_i[r] - mn);
      const float p0 = __expf(s0[r] - mn);
      const float p1 = __expf(s1[r] - mn);
      float rs = p0 + p1;
      for (int off = 1; off < 16; off <<= 1) rs += __shfl_xor(rs, off, 32);
      l_i[r] = l_i[r] * alpha + rs;
      m_i[r] = mn;
#pragma unroll
      for (int nt = 0; nt < 4; ++nt) o[nt][r] *= alpha;
      pt[wid][r + 8 * half][nl]      = (__bf16)p0;  // reshape C-layout -> A-layout
      pt[wid][r + 8 * half][16 + nl] = (__bf16)p1;
    }
    const bf16x16 pa = ld_frag(&pt[wid][0][0], 40);  // P(16x32) as A operand
#pragma unroll
    for (int nt = 0; nt < 4; ++nt) {
      const bf16x16 vf = ld_frag(&vt[nt * 16][0], 40);  // V(32x16) as B operand
      o[nt] = wmma_bf16(pa, vf, o[nt]);
    }
  }
  // finalize: O /= l, write bf16 into [B*N, HD] with head-concat layout
#pragma unroll
  for (int nt = 0; nt < 4; ++nt) {
#pragma unroll
    for (int r = 0; r < 8; ++r) {
      const int i = qrow0 + r + 8 * half;
      const int d = nt * 16 + nl;
      const float v = o[nt][r] / l_i[r];
      out[(size_t)(b * NN + i) * HD + h * DHEAD + d] = (__bf16)v;
    }
  }
}

// ---------------- GLU activation: act = a * silu(g), bf16 for next GEMM ----------------
__global__ void glu_k(const float* __restrict__ p, __bf16* __restrict__ act) {
  int idx = blockIdx.x * blockDim.x + threadIdx.x;
  const int total = MTOT * FF_HID;
  for (; idx < total; idx += gridDim.x * blockDim.x) {
    const int row = idx / FF_HID, c = idx - row * FF_HID;
    const float* pr = p + (size_t)row * GLU_OUT;
    const float a = pr[c], g = pr[FF_HID + c];
    act[idx] = (__bf16)(a * (g / (1.f + __expf(-g))));
  }
}

// ---------------- host orchestration ----------------
extern "C" void kernel_launch(void* const* d_in, const int* in_sizes, int n_in,
                              void* d_out, int out_size, void* d_ws, size_t ws_size,
                              hipStream_t stream) {
  (void)in_sizes; (void)n_in; (void)out_size; (void)ws_size;
  const float*         x_in  = (const float*)d_in[0];
  const unsigned char* mask  = (const unsigned char*)d_in[1];
  const float* temperature   = (const float*)d_in[2];
  const float* pw1   = (const float*)d_in[3];
  const float* pb1   = (const float*)d_in[4];
  const float* pw2   = (const float*)d_in[5];
  const float* pb2   = (const float*)d_in[6];
  const float* pw3   = (const float*)d_in[7];
  const float* pb3   = (const float*)d_in[8];
  const float* ln1_g = (const float*)d_in[9];
  const float* ln1_b = (const float*)d_in[10];
  const float* qkv_w = (const float*)d_in[11];
  const float* out_w = (const float*)d_in[12];
  const float* ln2_g = (const float*)d_in[13];
  const float* ln2_b = (const float*)d_in[14];
  const float* glu_w = (const float*)d_in[15];
  const float* glu_b = (const float*)d_in[16];
  const float* ff2_w = (const float*)d_in[17];
  const float* ff2_b = (const float*)d_in[18];

  char* ws = (char*)d_ws;
  size_t off = 0;
  auto alloc = [&](size_t bytes) -> void* {
    void* p = ws + off;
    off = (off + bytes + 255) & ~(size_t)255;
    return p;
  };
  float*  x    = (float*)alloc((size_t)MTOT * DIMM * 4);
  __bf16* xn   = (__bf16*)alloc((size_t)MTOT * DIMM * 2);
  float*  qkv  = (float*)alloc((size_t)MTOT * 3 * HD * 4);
  __bf16* qb   = (__bf16*)alloc((size_t)BB * NHEAD * NN * DHEAD * 2);
  __bf16* kb   = (__bf16*)alloc((size_t)BB * NHEAD * NN * DHEAD * 2);
  __bf16* vb   = (__bf16*)alloc((size_t)BB * NHEAD * NN * DHEAD * 2);
  __bf16* act  = (__bf16*)alloc((size_t)MTOT * FF_HID * 2);  // attn-out (ld HD) / glu-act (ld FF_HID)
  float*  p    = (float*)alloc((size_t)MTOT * GLU_OUT * 4);
  float*  htab = (float*)alloc((size_t)(2 * NN - 1) * NHEAD * 4);
  float*  tmpv = (float*)alloc(BB * 4);

  hipMemcpyAsync(x, x_in, (size_t)MTOT * DIMM * 4, hipMemcpyDeviceToDevice, stream);
  posmlp_k<<<2 * NN - 1, PDIM, 0, stream>>>(pw1, pb1, pw2, pb2, pw3, pb3, htab);
  temp_k<<<1, 256, 0, stream>>>(mask, temperature, tmpv);

  for (int i = 0; i < DEPTH; ++i) {
    ln_k<<<MTOT, 256, 0, stream>>>(x, ln1_g + i * DIMM, ln1_b + i * DIMM, xn);
    gemm_bf16_k<0><<<dim3((3 * HD) / 64, MTOT / 128), 256, 0, stream>>>(
        xn, DIMM, qkv_w + (size_t)i * DIMM * 3 * HD, 3 * HD, qkv, 3 * HD,
        nullptr, nullptr, DIMM);
    qkvprep_k<<<MTOT, 384, 0, stream>>>(qkv, qb, kb, vb);
    attn_k<<<dim3(NN / 64, BB * NHEAD), 128, 0, stream>>>(qb, kb, vb, mask, tmpv,
                                                          htab, act);
    gemm_bf16_k<2><<<dim3(DIMM / 64, MTOT / 128), 256, 0, stream>>>(
        act, HD, out_w + (size_t)i * HD * DIMM, DIMM, x, DIMM, nullptr, x, HD);
    ln_k<<<MTOT, 256, 0, stream>>>(x, ln2_g + i * DIMM, ln2_b + i * DIMM, xn);
    gemm_bf16_k<1><<<dim3(GLU_OUT / 64, MTOT / 128), 256, 0, stream>>>(
        xn, DIMM, glu_w + (size_t)i * DIMM * GLU_OUT, GLU_OUT, p, GLU_OUT,
        glu_b + (size_t)i * GLU_OUT, nullptr, DIMM);
    glu_k<<<1024, 256, 0, stream>>>(p, act);
    gemm_bf16_k<3><<<dim3(DIMM / 64, MTOT / 128), 256, 0, stream>>>(
        act, FF_HID, ff2_w + (size_t)i * FF_HID * DIMM, DIMM, x, DIMM,
        ff2_b + (size_t)i * DIMM, x, FF_HID);
  }
  hipMemcpyAsync(d_out, x, (size_t)MTOT * DIMM * 4, hipMemcpyDeviceToDevice, stream);
}
